// NeuralODE_82892868813103
// MI455X (gfx1250) — matrix operand order repaired and checked by reading, not compiled, not run on hardware
//
#include <hip/hip_runtime.h>
#include <hip/hip_bf16.h>

// ---------------------------------------------------------------------------
// Neural ODE (RK3 scan) on MI455X / gfx1250.
// All GEMMs via v_wmma_f32_16x16x32_bf16 (bf16 A/B, fp32 accumulate).
// Weights pre-swizzled once into a lane-major B-fragment layout, contiguous k (bf16, in ws).
// State + activations live in LDS; only y0 (in) and ys (out) touch HBM.
// B fragments are re-streamed from L2 every step through an opaque
// address_space(1) pointer: blocks LICM hoisting/spilling (round-1 bug) while
// still lowering to global_load_b128 (round-2 emitted flat_load_b128).
// ---------------------------------------------------------------------------

typedef __attribute__((ext_vector_type(16))) __bf16 v16bf;
typedef __attribute__((ext_vector_type(8)))  __bf16 v8bf;
typedef __attribute__((ext_vector_type(8)))  float  v8f;

typedef __bf16 __attribute__((address_space(1))) gbf16;   // global-AS bf16
typedef const __attribute__((address_space(1))) v8bf* gv8bfp;

#define TT 256
#define BB 4096
#define DD 64

#define MT    32            // batch rows per workgroup
#define NWAVE 4
#define NTHR  (NWAVE * 32)

// packed bf16 weight offsets (elements) inside d_ws
#define P0_OFF 0                       // layer0: K=64,  N=256 -> 2*16 tiles
#define P1_OFF 16384                   // layer1: K=256, N=256 -> 8*16 tiles
#define P2_OFF (16384 + 65536)         // layer2
#define P3_OFF (16384 + 2 * 65536)     // layer3: K=256, N=64  -> 8*4 tiles
#define PTOT   163840                  // total packed bf16 elements (320 KB)

// LDS row strides (elements), padded so 16-row b128 loads hit distinct banks
#define S0 72     // X0 (bf16, 32x64 input staging)
#define SH 264    // H0/H1 (bf16, 32x256 hidden staging)
#define SY 68     // ybuf/abuf (f32)

// ---------------------------------------------------------------------------
// Weight pre-pack: fp32 [N][K] row-major -> bf16 B-fragments.
// Tile (kt,nt) holds a 32x16 (KxN) fragment, 512 bf16, laid out lane-major:
//   element e of lane L  <->  n = nt*16 + (L&15),  k = kt*32 + (L>>4)*16 + e
// contiguous k order, the same order load_a uses for the A fragment.
// ---------------------------------------------------------------------------
__global__ void pack_weights_kernel(const float* __restrict__ w0,
                                    const float* __restrict__ w1,
                                    const float* __restrict__ w2,
                                    const float* __restrict__ w3,
                                    __bf16* __restrict__ pk) {
    int gid = blockIdx.x * 256 + threadIdx.x;
    if (gid >= PTOT) return;
    const float* w; int K, N, base;
    if (gid < P1_OFF)      { w = w0; K = 64;  N = 256; base = P0_OFF; }
    else if (gid < P2_OFF) { w = w1; K = 256; N = 256; base = P1_OFF; }
    else if (gid < P3_OFF) { w = w2; K = 256; N = 256; base = P2_OFF; }
    else                   { w = w3; K = 256; N = 64;  base = P3_OFF; }
    int r    = gid - base;
    int tile = r >> 9;            // /512 elements per tile
    int idx  = r & 511;
    int lane = idx >> 4;
    int e    = idx & 15;
    int ntn  = N >> 4;
    int kt   = tile / ntn;
    int nt   = tile % ntn;
    int n = nt * 16 + (lane & 15);
    int k = kt * 32 + ((lane >> 4) << 4) + e;
    pk[gid] = (__bf16)w[n * K + k];
}

// ---------------------------------------------------------------------------
// Fragment loaders
// ---------------------------------------------------------------------------

// A fragment (16x32 MxK, 16-bit layout): lane<16 holds row M=lane, K={0..15};
// lane>=16 holds row M=lane-16, K={16..31}. Two ds_load_b128 per lane.
__device__ __forceinline__ v16bf load_a(const __bf16* X, int stride, int m0,
                                        int kt, int lane) {
    int r  = m0 + (lane & 15);
    int c0 = kt * 32 + ((lane >> 4) << 4);
    const __bf16* p = X + r * stride + c0;
    union { v16bf f; v8bf h[2]; } u;
    u.h[0] = *(const v8bf*)(p);
    u.h[1] = *(const v8bf*)(p + 8);
    return u.f;
}

// B fragment from pre-swizzled global weights: two coalesced global_load_b128.
__device__ __forceinline__ v16bf load_b(const gbf16* Wp, int tile, int lane) {
    const gbf16* p = Wp + tile * 512 + lane * 16;
    union { v16bf f; v8bf h[2]; } u;
    u.h[0] = *(gv8bfp)(p);
    u.h[1] = *(gv8bfp)(p + 8);
    return u.f;
}

// Opaque global-AS weight pointer: the empty asm makes it loop-variant (no LICM
// hoist/spill across the scan), the address_space(1) type keeps loads "global".
__device__ __forceinline__ const gbf16* opaque_g(const __bf16* p) {
    const gbf16* q = (const gbf16*)p;
    asm volatile("" : "+s"(q));
    return q;
}

__device__ __forceinline__ float softplus_f(float x) {
    // stable softplus: max(x,0) + log1p(exp(-|x|)); fast trans ops
    float ax = __builtin_fabsf(x);
    return fmaxf(x, 0.0f) + __logf(1.0f + __expf(-ax));
}

// ---------------------------------------------------------------------------
// Hidden layer: out[MTx256] = softplus(in[MTxK] @ W^T + b), bf16 in/out in LDS.
// Wave owns a 16-row x 128-col stripe: 8 fp32 accumulator tiles.
// Accumulators start at inline 0 (folds into first WMMA's SRC2); bias is added
// in the epilogue where it fuses into the softplus chain.
// Caller provides __syncthreads() between layers (ping-pong buffers).
// ---------------------------------------------------------------------------
template <int KDIM>
__device__ __forceinline__ void layer_hidden(const __bf16* __restrict__ Xin, int sin,
                                             __bf16* __restrict__ Xout,
                                             const __bf16* __restrict__ Wp,
                                             const float* __restrict__ bias,
                                             int m0, int n0, int lane) {
    const gbf16* W = opaque_g(Wp);

    v8f acc[8];
#pragma unroll
    for (int j = 0; j < 8; ++j)
        acc[j] = (v8f){0.f, 0.f, 0.f, 0.f, 0.f, 0.f, 0.f, 0.f};

#pragma unroll
    for (int kt = 0; kt < KDIM / 32; ++kt) {
        v16bf a = load_a(Xin, sin, m0, kt, lane);
#pragma unroll
        for (int j = 0; j < 8; ++j) {
            v16bf b = load_b(W, kt * 16 + (n0 >> 4) + j, lane);
            acc[j] = __builtin_amdgcn_wmma_f32_16x16x32_bf16(
                false, a, false, b, (short)0, acc[j], false, false);
        }
    }
    // bias + softplus in registers, then bf16 store to LDS staging
    int hi = lane >> 4;
#pragma unroll
    for (int j = 0; j < 8; ++j) {
        int n = n0 + j * 16 + (lane & 15);
        float bs = bias[n];
#pragma unroll
        for (int v = 0; v < 8; ++v) {
            int r = m0 + v + hi * 8;  // C/D layout: VGPR v = row (lanes>=16 -> +8)
            Xout[r * SH + n] = (__bf16)softplus_f(acc[j][v] + bs);
        }
    }
}

// Final layer (N=64): each wave owns 2 output tiles; result stays in VGPRs.
__device__ __forceinline__ void layer_final(const __bf16* __restrict__ Xin,
                                            const __bf16* __restrict__ Wp,
                                            int m0, int ntb, int lane, v8f out[2]) {
    const gbf16* W = opaque_g(Wp);
    out[0] = (v8f){0.f, 0.f, 0.f, 0.f, 0.f, 0.f, 0.f, 0.f};
    out[1] = (v8f){0.f, 0.f, 0.f, 0.f, 0.f, 0.f, 0.f, 0.f};
#pragma unroll
    for (int kt = 0; kt < 8; ++kt) {
        v16bf a = load_a(Xin, SH, m0, kt, lane);
#pragma unroll
        for (int j = 0; j < 2; ++j) {
            v16bf b = load_b(W, kt * 4 + ntb + j, lane);
            out[j] = __builtin_amdgcn_wmma_f32_16x16x32_bf16(
                false, a, false, b, (short)0, out[j], false, false);
        }
    }
}

// ---------------------------------------------------------------------------
// Main kernel: one workgroup = 32 batch rows, full 256-step RK3 scan on-chip.
// ---------------------------------------------------------------------------
__global__ __launch_bounds__(NTHR) void node_scan_kernel(
    const float* __restrict__ ts, const float* __restrict__ y0,
    const float* __restrict__ b0, const float* __restrict__ b1,
    const float* __restrict__ b2, const float* __restrict__ b3,
    const __bf16* __restrict__ pk, float* __restrict__ out) {

    __shared__ float  ybuf[MT * SY];     // current state y (fp32)
    __shared__ float  abuf[MT * SY];     // RK accumulator  (fp32)
    __shared__ __bf16 X0[MT * S0];       // MLP input staging (32x64)
    __shared__ __bf16 H0[MT * SH];       // hidden staging ping
    __shared__ __bf16 H1[MT * SH];       // hidden staging pong
    __shared__ float  sb0[256], sb1[256], sb2[256], sb3[64];

    const int tid  = threadIdx.x;
    const int lane = tid & 31;
    const int w    = tid >> 5;
    const int blk  = blockIdx.x;

    const int m0  = (w & 1) * 16;        // wave's row tile inside the 32-row block
    const int n0  = (w >> 1) * 128;      // wave's column stripe for hidden layers
    const int ntb = (w >> 1) * 2;        // wave's tile base for final layer (N=64)

    const float dtv = ts[1] - ts[0];

    for (int i = tid; i < 256; i += NTHR) { sb0[i] = b0[i]; sb1[i] = b1[i]; sb2[i] = b2[i]; }
    for (int i = tid; i < 64;  i += NTHR) sb3[i] = b3[i];

    const float* ysrc = y0 + (size_t)blk * MT * DD;
    for (int i = tid; i < MT * DD; i += NTHR) {
        int r = i >> 6, c = i & 63;
        float v = ysrc[i];
        ybuf[r * SY + c] = v;
        X0[r * S0 + c]   = (__bf16)v;
    }

#pragma unroll 1
    for (int t = 0; t < TT; ++t) {
#pragma unroll 1
        for (int p = 0; p < 3; ++p) {
            __syncthreads();  // X0 ready; previous consumers of H0 done
            layer_hidden<64 >(X0, S0, H0, pk + P0_OFF, sb0, m0, n0, lane);
            __syncthreads();
            layer_hidden<256>(H0, SH, H1, pk + P1_OFF, sb1, m0, n0, lane);
            __syncthreads();
            layer_hidden<256>(H1, SH, H0, pk + P2_OFF, sb2, m0, n0, lane);
            __syncthreads();
            v8f kk[2];
            layer_final(H0, pk + P3_OFF, m0, ntb, lane, kk);

            // RK3 epilogue; each (r,n) slot is owned by this lane in all phases.
            const int hi = lane >> 4;
#pragma unroll
            for (int j = 0; j < 2; ++j) {
                int n = (ntb + j) * 16 + (lane & 15);
                float bs = sb3[n];
#pragma unroll
                for (int v = 0; v < 8; ++v) {
                    int r = m0 + v + hi * 8;
                    float kv = kk[j][v] + bs;            // k1/k2/k3 element
                    float y  = ybuf[r * SY + n];
                    if (p == 0) {
                        abuf[r * SY + n] = (2.0f / 9.0f) * kv;
                        X0[r * S0 + n]   = (__bf16)(y + 0.5f * kv);
                    } else if (p == 1) {
                        abuf[r * SY + n] += (1.0f / 3.0f) * kv;
                        X0[r * S0 + n]    = (__bf16)(y + 0.75f * kv);
                    } else {
                        float y1 = y + dtv * (abuf[r * SY + n] + (4.0f / 9.0f) * kv);
                        ybuf[r * SY + n] = y1;
                        X0[r * S0 + n]   = (__bf16)y1;   // input for next step's k1
                    }
                }
            }
        }
        __syncthreads();  // ybuf fully updated
        float* od = out + (size_t)t * BB * DD + (size_t)blk * MT * DD;
        for (int i = tid; i < MT * DD; i += NTHR) {
            int r = i >> 6, c = i & 63;
            od[i] = ybuf[r * SY + c];
        }
    }
}

// ---------------------------------------------------------------------------
extern "C" void kernel_launch(void* const* d_in, const int* in_sizes, int n_in,
                              void* d_out, int out_size, void* d_ws, size_t ws_size,
                              hipStream_t stream) {
    (void)in_sizes; (void)n_in; (void)out_size; (void)ws_size;
    const float* ts = (const float*)d_in[0];
    const float* y0 = (const float*)d_in[1];
    const float* w0 = (const float*)d_in[2];
    const float* b0 = (const float*)d_in[3];
    const float* w1 = (const float*)d_in[4];
    const float* b1 = (const float*)d_in[5];
    const float* w2 = (const float*)d_in[6];
    const float* b2 = (const float*)d_in[7];
    const float* w3 = (const float*)d_in[8];
    const float* b3 = (const float*)d_in[9];

    __bf16* pk = (__bf16*)d_ws;  // 320 KB of packed bf16 weights

    pack_weights_kernel<<<(PTOT + 255) / 256, 256, 0, stream>>>(w0, w1, w2, w3, pk);
    node_scan_kernel<<<BB / MT, NTHR, 0, stream>>>(ts, y0, b0, b1, b2, b3, pk,
                                                   (float*)d_out);
}
